// VisionMamba_24859270709412
// MI455X (gfx1250) — matrix operand order repaired and checked by reading, not compiled, not run on hardware
//
#include <hip/hip_runtime.h>
#include <hip/hip_bf16.h>

// ---------- model dims ----------
#define BSZ     8
#define LSEQ    196
#define DMODEL  768
#define DIN     1536      // DI
#define DSN     16        // DS
#define DTRK    48        // DTR
#define DTRP    64        // DTR padded to multiple of 32 for WMMA K
#define NLAY    12
#define MROWS   (BSZ*LSEQ)   // 1568 = 49*32
#define NPROJ   80           // DTR + 2*DS
#define NPROJP  96           // NPROJ padded to multiple of 32 for WMMA N

typedef __attribute__((ext_vector_type(16))) __bf16 v16bf;
typedef __attribute__((ext_vector_type(8)))  __bf16 v8bf;
typedef __attribute__((ext_vector_type(8)))  float  v8f;

union AV { v16bf v; v8bf h[2]; };

__device__ __forceinline__ unsigned short f2bf(float f) {
  unsigned u = __float_as_uint(f);
  u += 0x7FFFu + ((u >> 16) & 1u);     // round-to-nearest-even
  return (unsigned short)(u >> 16);
}

#define WMMA_BF16(a, b, c) \
  __builtin_amdgcn_wmma_f32_16x16x32_bf16(false, (a), false, (b), (short)0, (c), false, false)

// ---------------------------------------------------------------------------
// bf16 WMMA GEMM: C[M x N] = A[M x K] * W[N x K]^T, f32 accumulate.
// One wave per 32(M) x 32(N) tile: 4 accumulators, 2 A-frags x 2 B-frags per
// K-step (4 v_wmma_f32_16x16x32_bf16). Software-pipelined: K-block k+32 is
// loaded into a second register set before the WMMAs for block k issue, so
// s_wait_loadcnt only drains older loads and L2 latency overlaps matrix math.
// Requires K % 32 == 0, M % 32 == 0, N % 32 == 0.  grid = (N/32, M/32).
// ---------------------------------------------------------------------------
__global__ __launch_bounds__(32)
void k_gemm(const unsigned short* __restrict__ A,
            const unsigned short* __restrict__ W,
            float* __restrict__ C, int N, int K) {
  const int lane = threadIdx.x;
  const int half = lane >> 4;      // 0: lanes 0-15, 1: lanes 16-31
  const int lid  = lane & 15;
  const int n0 = blockIdx.x << 5;
  const int m0 = blockIdx.y << 5;

  const unsigned short* arow0 = A + (size_t)(m0 + lid) * K;      // rows m0..m0+15
  const unsigned short* arow1 = arow0 + (size_t)16 * K;          // rows m0+16..
  const unsigned short* brow0 = W + (size_t)(n0 + lid) * K + half * 16; // cols n0..
  const unsigned short* brow1 = brow0 + (size_t)16 * K;          // cols n0+16..
  const int ka = half * 16;

  v8f c00 = {}, c01 = {}, c10 = {}, c11 = {};

  // ---- prologue: preload K-block 0 ----
  AV a0, a1; v16bf b0, b1;
  a0.h[0] = *(const v8bf*)(arow0 + ka);
  a0.h[1] = *(const v8bf*)(arow0 + ka + 8);
  a1.h[0] = *(const v8bf*)(arow1 + ka);
  a1.h[1] = *(const v8bf*)(arow1 + ka + 8);
  b0 = *(const v16bf*)(brow0);
  b1 = *(const v16bf*)(brow1);

  for (int kb = 32; kb < K; kb += 32) {
    // issue next K-block's loads first (independent registers)
    AV na0, na1; v16bf nb0, nb1;
    na0.h[0] = *(const v8bf*)(arow0 + kb + ka);
    na0.h[1] = *(const v8bf*)(arow0 + kb + ka + 8);
    na1.h[0] = *(const v8bf*)(arow1 + kb + ka);
    na1.h[1] = *(const v8bf*)(arow1 + kb + ka + 8);
    nb0 = *(const v16bf*)(brow0 + kb);
    nb1 = *(const v16bf*)(brow1 + kb);
    if (kb + 32 < K) {
      __builtin_prefetch(arow0 + kb + 32, 0, 3);  // global_prefetch_b8
      __builtin_prefetch(brow0 + kb + 32, 0, 3);
    }
    // WMMAs on current block while next block's loads are in flight
    c00 = WMMA_BF16(a0.v, b0, c00);
    c01 = WMMA_BF16(a0.v, b1, c01);
    c10 = WMMA_BF16(a1.v, b0, c10);
    c11 = WMMA_BF16(a1.v, b1, c11);
    a0 = na0; a1 = na1; b0 = nb0; b1 = nb1;
  }
  // ---- epilogue: last K-block ----
  c00 = WMMA_BF16(a0.v, b0, c00);
  c01 = WMMA_BF16(a0.v, b1, c01);
  c10 = WMMA_BF16(a1.v, b0, c10);
  c11 = WMMA_BF16(a1.v, b1, c11);

  // C/D layout: VGPR r -> row (r + 8*half), col = n0 + lid (+16 for right tiles)
  float* cp = C + (size_t)(m0 + half * 8) * N + n0 + lid;
  float* cq = cp + (size_t)16 * N;
#pragma unroll
  for (int r = 0; r < 8; ++r) {
    cp[(size_t)r * N]      = c00[r];
    cp[(size_t)r * N + 16] = c01[r];
    cq[(size_t)r * N]      = c10[r];
    cq[(size_t)r * N + 16] = c11[r];
  }
}

// ---------------------------------------------------------------------------
// elementwise / prep kernels
// ---------------------------------------------------------------------------
__global__ void k_cvt(const float* __restrict__ s, unsigned short* __restrict__ d, int n) {
  int i = blockIdx.x * blockDim.x + threadIdx.x;
  if (i < n) d[i] = f2bf(s[i]);
}

// pad columns kin -> kout (zeros beyond kin); src row stride = sstride
__global__ void k_pad_bf16(const float* __restrict__ s, unsigned short* __restrict__ d,
                           int rows, int kin, int kout, int sstride) {
  int i = blockIdx.x * blockDim.x + threadIdx.x;
  if (i >= rows * kout) return;
  int r = i / kout, k = i - r * kout;
  d[i] = (k < kin) ? f2bf(s[(size_t)r * sstride + k]) : (unsigned short)0;
}

// pad rows rin -> rout (zero rows beyond rin), K columns dense
__global__ void k_pad_rows_bf16(const float* __restrict__ s, unsigned short* __restrict__ d,
                                int rin, int rout, int K) {
  int i = blockIdx.x * blockDim.x + threadIdx.x;
  if (i >= rout * K) return;
  int r = i / K;
  d[i] = (r < rin) ? f2bf(s[i]) : (unsigned short)0;
}

// rearrange image into patch rows (bf16): row m=b*196+n, feature f=c*256+p*16+q
__global__ void k_patch_prep(const float* __restrict__ x, unsigned short* __restrict__ d) {
  int i = blockIdx.x * blockDim.x + threadIdx.x;
  if (i >= MROWS * DMODEL) return;
  int m = i / DMODEL, f = i - m * DMODEL;
  int b = m / LSEQ, n = m - b * LSEQ;
  int pi = n / 14, pj = n - pi * 14;
  int c = f >> 8, p = (f >> 4) & 15, q = f & 15;
  float v = x[(((size_t)b * 3 + c) * 224 + (pi * 16 + p)) * 224 + (pj * 16 + q)];
  d[i] = f2bf(v);
}

__global__ void k_bias_pos(float* __restrict__ h, const float* __restrict__ pb,
                           const float* __restrict__ pe) {
  int i = blockIdx.x * blockDim.x + threadIdx.x;
  if (i >= MROWS * DMODEL) return;
  int m = i / DMODEL, dd = i - m * DMODEL, n = m % LSEQ;
  h[i] += pb[dd] + pe[(size_t)n * DMODEL + dd];
}

// causal depthwise conv (k=4) + SiLU; writes f32 (for scan) and bf16 (for x_proj GEMM)
__global__ void k_conv_silu(const float* __restrict__ xz, const float* __restrict__ cw,
                            const float* __restrict__ cb, float* __restrict__ xc,
                            unsigned short* __restrict__ xcb) {
  int i = blockIdx.x * blockDim.x + threadIdx.x;
  if (i >= MROWS * DIN) return;
  int m = i / DIN, e = i - m * DIN;
  int t = m % LSEQ, mb = m - t;
  float acc = cb[e];
#pragma unroll
  for (int k = 0; k < 4; ++k) {
    int st = t - 3 + k;
    if (st >= 0) acc += xz[(size_t)(mb + st) * (2 * DIN) + e] * cw[e * 4 + k];
  }
  float s = acc / (1.f + __expf(-acc));
  xc[i]  = s;
  xcb[i] = f2bf(s);
}

__global__ void k_softplus(float* __restrict__ dt, const float* __restrict__ dtb) {
  int i = blockIdx.x * blockDim.x + threadIdx.x;
  if (i >= MROWS * DIN) return;
  float v = dt[i] + dtb[i % DIN];
  dt[i] = (v > 20.f) ? v : log1pf(__expf(v));
}

// selective scan: one thread per (batch, channel); 16 f32 states in registers.
// Fuses +x*D, SiLU(z) gate, and bf16 output for the out_proj GEMM.
__global__ void k_scan(const float* __restrict__ xc, const float* __restrict__ dt,
                       const float* __restrict__ proj, const float* __restrict__ xz,
                       const float* __restrict__ alog, const float* __restrict__ dpar,
                       unsigned short* __restrict__ ybf) {
  int idx = blockIdx.x * blockDim.x + threadIdx.x;   // exactly BSZ*DIN threads
  int b = idx / DIN, e = idx - b * DIN;
  float A[DSN], h[DSN];
#pragma unroll
  for (int s = 0; s < DSN; ++s) { A[s] = -__expf(alog[(size_t)e * DSN + s]); h[s] = 0.f; }
  const float Dp = dpar[e];
  for (int t = 0; t < LSEQ; ++t) {
    size_t m = (size_t)b * LSEQ + t;
    float xv = xc[m * DIN + e];
    float d  = dt[m * DIN + e];
    float dx = d * xv;
    const float* pr = proj + m * NPROJP;
    float y = 0.f;
#pragma unroll
    for (int s = 0; s < DSN; ++s) {
      h[s] = __expf(d * A[s]) * h[s] + dx * pr[DTRK + s];
      y += h[s] * pr[DTRK + DSN + s];
    }
    float z = xz[m * (2 * DIN) + DIN + e];
    float out = (y + xv * Dp) * (z / (1.f + __expf(-z)));
    ybf[m * DIN + e] = f2bf(out);
  }
}

__global__ void k_layernorm(const float* __restrict__ h, const float* __restrict__ nw,
                            const float* __restrict__ nb, float* __restrict__ out) {
  __shared__ float red[256];
  int m = blockIdx.x, tid = threadIdx.x;
  const float* row = h + (size_t)m * DMODEL;
  float s = 0.f;
  for (int d = tid; d < DMODEL; d += 256) s += row[d];
  red[tid] = s; __syncthreads();
  for (int o = 128; o > 0; o >>= 1) { if (tid < o) red[tid] += red[tid + o]; __syncthreads(); }
  float mu = red[0] * (1.f / DMODEL); __syncthreads();
  float v = 0.f;
  for (int d = tid; d < DMODEL; d += 256) { float t = row[d] - mu; v += t * t; }
  red[tid] = v; __syncthreads();
  for (int o = 128; o > 0; o >>= 1) { if (tid < o) red[tid] += red[tid + o]; __syncthreads(); }
  float inv = rsqrtf(red[0] * (1.f / DMODEL) + 1e-5f);
  for (int d = tid; d < DMODEL; d += 256)
    out[(size_t)m * DMODEL + d] = (row[d] - mu) * inv * nw[d] + nb[d];
}

// ---------------------------------------------------------------------------
extern "C" void kernel_launch(void* const* d_in, const int* in_sizes, int n_in,
                              void* d_out, int out_size, void* d_ws, size_t ws_size,
                              hipStream_t stream) {
  (void)in_sizes; (void)n_in; (void)out_size; (void)ws_size;
  const float* x         = (const float*)d_in[0];
  const float* patch_w   = (const float*)d_in[1];
  const float* patch_b   = (const float*)d_in[2];
  const float* pos_embed = (const float*)d_in[3];
  const float* in_proj_w = (const float*)d_in[4];
  const float* conv_w    = (const float*)d_in[5];
  const float* conv_b    = (const float*)d_in[6];
  const float* x_proj_w  = (const float*)d_in[7];
  const float* dt_proj_w = (const float*)d_in[8];
  const float* dt_proj_b = (const float*)d_in[9];
  const float* A_log     = (const float*)d_in[10];
  const float* D_param   = (const float*)d_in[11];
  const float* out_pw    = (const float*)d_in[12];
  const float* norm_w    = (const float*)d_in[13];
  const float* norm_b    = (const float*)d_in[14];

  // workspace layout (all 256B aligned); total ~53.5 MB
  char* ws = (char*)d_ws;
  float*          hbuf = (float*)(ws + 0);                       //  4,816,896 B (1568x768 f32)
  float*          xz   = (float*)(ws + 4816896);                 // 19,267,584 B (1568x3072 f32)
  float*          xc   = (float*)(ws + 24084480);                //  9,633,792 B (1568x1536 f32)
  float*          pj   = (float*)(ws + 33718272);                //    602,112 B (1568x96 f32)
  float*          dtf  = (float*)(ws + 34320384);                //  9,633,792 B (1568x1536 f32)
  unsigned short* abf  = (unsigned short*)(ws + 43954176);       //  4,816,896 B (staged bf16 acts)
  unsigned short* wbf  = (unsigned short*)(ws + 48771072);       //  4,718,592 B (staged bf16 weights)

  auto NB = [](int n) { return (n + 255) / 256; };
  const dim3 B256(256);

  // ---- patch embedding ----
  k_patch_prep<<<NB(MROWS * DMODEL), B256, 0, stream>>>(x, abf);
  k_cvt<<<NB(DMODEL * DMODEL), B256, 0, stream>>>(patch_w, wbf, DMODEL * DMODEL);
  k_gemm<<<dim3(DMODEL / 32, MROWS / 32), 32, 0, stream>>>(abf, wbf, hbuf, DMODEL, DMODEL);
  k_bias_pos<<<NB(MROWS * DMODEL), B256, 0, stream>>>(hbuf, patch_b, pos_embed);

  // ---- mamba layers ----
  for (int l = 0; l < NLAY; ++l) {
    // in_proj: xz = h @ Win^T   (K=768, N=3072)
    k_cvt<<<NB(MROWS * DMODEL), B256, 0, stream>>>(hbuf, abf, MROWS * DMODEL);
    k_cvt<<<NB(2 * DIN * DMODEL), B256, 0, stream>>>(in_proj_w + (size_t)l * 2 * DIN * DMODEL,
                                                     wbf, 2 * DIN * DMODEL);
    k_gemm<<<dim3(2 * DIN / 32, MROWS / 32), 32, 0, stream>>>(abf, wbf, xz, 2 * DIN, DMODEL);

    // causal conv + SiLU
    k_conv_silu<<<NB(MROWS * DIN), B256, 0, stream>>>(xz, conv_w + (size_t)l * DIN * 4,
                                                      conv_b + (size_t)l * DIN, xc, abf);

    // x_proj: proj = xc @ Wxp^T  (K=1536, N padded 80->96 with zero weight rows)
    k_pad_rows_bf16<<<NB(NPROJP * DIN), B256, 0, stream>>>(x_proj_w + (size_t)l * NPROJ * DIN,
                                                           wbf, NPROJ, NPROJP, DIN);
    k_gemm<<<dim3(NPROJP / 32, MROWS / 32), 32, 0, stream>>>(abf, wbf, pj, NPROJP, DIN);

    // dt_proj: dt = proj[:, :48] @ Wdt^T  (K padded 48->64, N=1536)
    k_pad_bf16<<<NB(MROWS * DTRP), B256, 0, stream>>>(pj, abf, MROWS, DTRK, DTRP, NPROJP);
    k_pad_bf16<<<NB(DIN * DTRP), B256, 0, stream>>>(dt_proj_w + (size_t)l * DIN * DTRK,
                                                    wbf, DIN, DTRK, DTRP, DTRK);
    k_gemm<<<dim3(DIN / 32, MROWS / 32), 32, 0, stream>>>(abf, wbf, dtf, DIN, DTRP);
    k_softplus<<<NB(MROWS * DIN), B256, 0, stream>>>(dtf, dt_proj_b + (size_t)l * DIN);

    // selective scan (+x*D, *silu(z)), emits bf16 y into abf
    k_scan<<<(BSZ * DIN) / 256, B256, 0, stream>>>(xc, dtf, pj, xz,
                                                   A_log + (size_t)l * DIN * DSN,
                                                   D_param + (size_t)l * DIN, abf);

    // out_proj: h = y @ Wout^T  (K=1536, N=768)
    k_cvt<<<NB(DMODEL * DIN), B256, 0, stream>>>(out_pw + (size_t)l * DMODEL * DIN, wbf, DMODEL * DIN);
    k_gemm<<<dim3(DMODEL / 32, MROWS / 32), 32, 0, stream>>>(abf, wbf, hbuf, DMODEL, DIN);
  }

  // ---- final layernorm ----
  k_layernorm<<<MROWS, B256, 0, stream>>>(hbuf, norm_w, norm_b, (float*)d_out);
}
